// RLIFLayer_89618787598524
// MI455X (gfx1250) — matrix operand order repaired and checked by reading, not compiled, not run on hardware
//
#include <hip/hip_runtime.h>

typedef __attribute__((ext_vector_type(16))) __bf16 v16bf;
typedef __attribute__((ext_vector_type(8)))  float  v8f;

namespace {
constexpr int BATCH  = 64;
constexpr int TSTEPS = 1000;
constexpr int HDIM   = 1024;   // H (output features)
constexpr int KDIM   = 1024;   // I (input features)
constexpr float ALPHA_LO = 0.8187307530779818f;   // exp(-1/5)
constexpr float ALPHA_HI = 0.9607894391523232f;   // exp(-1/25)

// workspace layout (bytes)
constexpr size_t WS_VZT   = 0;                      // uint32[1024*512] : Vz^T bf16-pairs (2 MB)
constexpr size_t WS_MASKS = 2ull * 1024 * 1024;     // uint16[4 groups][2 parity][16 rows][64 words]
constexpr size_t WS_CTR   = WS_MASKS + 16 * 1024;   // uint32[4] barrier counters
constexpr size_t WS_WX    = 4ull * 1024 * 1024;     // float[64000*1024] (~250 MB)
}

__device__ __forceinline__ unsigned short f2bf(float f) {
  unsigned u = __float_as_uint(f);
  u += 0x7FFFu + ((u >> 16) & 1u);         // round-to-nearest-even
  return (unsigned short)(u >> 16);
}
__device__ __forceinline__ unsigned pack2(float lo, float hi) {
  return (unsigned)f2bf(lo) | ((unsigned)f2bf(hi) << 16);
}

union FragBF { v16bf v; unsigned u[8]; };

// ---------------------------------------------------------------------------
// Phase 0: Vz^T (zero-diag, bf16, column-of-V contiguous) + reset barrier ctrs
// vzt[h*512 + w] packs (V[2w][h], V[2w+1][h]) with diagonal zeroed.
// ---------------------------------------------------------------------------
__global__ __launch_bounds__(256) void rlif_prep(const float* __restrict__ V,
                                                 unsigned* __restrict__ vzt,
                                                 unsigned* __restrict__ ctr) {
  int idx = blockIdx.x * 256 + threadIdx.x;        // 0 .. 1024*512-1
  if (idx < 4) ctr[idx] = 0u;
  int h = idx >> 9;
  int w = idx & 511;
  int j0 = 2 * w, j1 = 2 * w + 1;
  float v0 = (j0 == h) ? 0.f : V[(size_t)j0 * HDIM + h];
  float v1 = (j1 == h) ? 0.f : V[(size_t)j1 * HDIM + h];
  vzt[(size_t)h * 512 + w] = pack2(v0, v1);
}

// ---------------------------------------------------------------------------
// Phase 1: Wx = x @ W^T, bf16 WMMA, 128x128 tile per WG (8 waves).
// Double-buffered LDS stage (global fetch of chunk kc+1 issued before the 8
// WMMAs of chunk kc) and B-fragment LDS loads pipelined one WMMA ahead.
// Per-lane operand layouts (contiguous k, same order for A and B):
//   A 16x32: lane<16: K {k0..k0+15};              lane>=16: K {k0+16..k0+31}
//   B 32x16: lane<16: K {k0..k0+15};              lane>=16: K {k0+16..k0+31}
// ---------------------------------------------------------------------------
__global__ __launch_bounds__(256) void rlif_gemm(const float* __restrict__ X,
                                                 const float* __restrict__ W,
                                                 float* __restrict__ Wx) {
  __shared__ unsigned As[2][8][32][8];   // [buf][m-tile][lane][8 dwords = 16 bf16]
  __shared__ unsigned Bs[2][8][32][8];   // [buf][n-tile][lane][8 dwords]

  const int tid  = threadIdx.x;
  const int lane = tid & 31;
  const int wv   = tid >> 5;          // wave 0..7, owns m-tile row `wv`
  const int l16  = lane & 15;
  const int hi   = lane >> 4;
  const int m0   = blockIdx.x * 128;
  const int n0   = blockIdx.y * 128;

  const float* aptr = X + (size_t)(m0 + wv * 16 + l16) * KDIM + (hi ? 16 : 0);
  const float* bptr = W + (size_t)(n0 + wv * 16 + l16) * KDIM + (hi ? 16 : 0);

  float4 ra0, ra1, ra2, ra3, rb0, rb1, rb2, rb3;
  auto fetch = [&](int k0) {
    const float4* pa = (const float4*)(aptr + k0);
    ra0 = pa[0]; ra1 = pa[1];
    const float4* pa2 = (const float4*)(aptr + k0 + 8);
    ra2 = pa2[0]; ra3 = pa2[1];
    const float4* pb = (const float4*)(bptr + k0);
    rb0 = pb[0]; rb1 = pb[1]; rb2 = pb[2]; rb3 = pb[3];
  };
  auto stage = [&](int buf) {
    unsigned* da = As[buf][wv][lane];
    da[0] = pack2(ra0.x, ra0.y); da[1] = pack2(ra0.z, ra0.w);
    da[2] = pack2(ra1.x, ra1.y); da[3] = pack2(ra1.z, ra1.w);
    da[4] = pack2(ra2.x, ra2.y); da[5] = pack2(ra2.z, ra2.w);
    da[6] = pack2(ra3.x, ra3.y); da[7] = pack2(ra3.z, ra3.w);
    unsigned* db = Bs[buf][wv][lane];
    db[0] = pack2(rb0.x, rb0.y); db[1] = pack2(rb0.z, rb0.w);
    db[2] = pack2(rb1.x, rb1.y); db[3] = pack2(rb1.z, rb1.w);
    db[4] = pack2(rb2.x, rb2.y); db[5] = pack2(rb2.z, rb2.w);
    db[6] = pack2(rb3.x, rb3.y); db[7] = pack2(rb3.z, rb3.w);
  };

  v8f acc[8];
  for (int c = 0; c < 8; ++c)
    for (int i = 0; i < 8; ++i) acc[c][i] = 0.f;

  fetch(0);
  stage(0);
  __syncthreads();

  for (int kc = 0; kc < KDIM / 32; ++kc) {
    const int cur = kc & 1;
    if (kc + 1 < KDIM / 32) fetch((kc + 1) * 32);   // overlap with WMMAs below

    FragBF fa, fbc;
    for (int i = 0; i < 8; ++i) fa.u[i] = As[cur][wv][lane][i];
    for (int i = 0; i < 8; ++i) fbc.u[i] = Bs[cur][0][lane][i];
#pragma unroll
    for (int c = 0; c < 8; ++c) {
      FragBF fbn;
      if (c + 1 < 8)        // pipeline next B fragment ahead of this WMMA
        for (int i = 0; i < 8; ++i) fbn.u[i] = Bs[cur][c + 1][lane][i];
      acc[c] = __builtin_amdgcn_wmma_f32_16x16x32_bf16(
          false, fa.v, false, fbc.v, (short)0, acc[c], false, false);
      fbc = fbn;
    }

    if (kc + 1 < KDIM / 32) stage(1 - cur);
    __syncthreads();
  }

  // D layout: vgpr v -> row m = v + 8*hi, col n = l16
  const int mrow = m0 + wv * 16 + hi * 8;
  for (int c = 0; c < 8; ++c) {
    const int col = n0 + c * 16 + l16;
    for (int v = 0; v < 8; ++v)
      Wx[(size_t)(mrow + v) * HDIM + col] = acc[c][v];
  }
}

// ---------------------------------------------------------------------------
// Phase 2: persistent scan. 64 WGs = 4 batch-groups x 16 H-slices (64 cols).
// Each WG keeps its Vz^T slice in LDS (bf16, XOR-swizzled) for all 1000 steps.
// Spikes exchanged between the 16 WGs of a group as 2KB bitmasks per step,
// synchronized with a monotonic atomic-counter barrier. The per-step WMMA
// chain is split over two accumulators and software-pipelined.
// ---------------------------------------------------------------------------
__global__ __launch_bounds__(128) void rlif_scan(const float* __restrict__ alpha,
                                                 const float* __restrict__ ut0,
                                                 const float* __restrict__ st0,
                                                 const float* __restrict__ Wx,
                                                 const unsigned* __restrict__ vzt,
                                                 unsigned short* __restrict__ masks,
                                                 unsigned* __restrict__ ctr,
                                                 float* __restrict__ out) {
  __shared__ unsigned Vlds[64][512];      // 128 KB: column n, K pairs (XOR-swizzled)
  __shared__ unsigned Asw[32][32][8];     // 32 KB: [kstep][lane][8 dwords] A operand

  const int tid   = threadIdx.x;          // 0..127
  const int lane  = tid & 31;
  const int wv    = tid >> 5;             // 0..3 (one 16-col N-tile per wave)
  const int l16   = lane & 15;
  const int hi    = lane >> 4;
  const int g     = blockIdx.x >> 4;      // batch group 0..3
  const int slice = blockIdx.x & 15;      // H slice 0..15
  const int n0    = slice * 64;

  // ---- load Vz^T slice into LDS (once), XOR swizzle groups of 8 dwords ----
  for (int i = tid; i < 64 * 512; i += 128) {
    int n = i >> 9, w = i & 511;
    Vlds[n][w ^ ((n & 7) << 3)] = vzt[(size_t)(n0 + n) * 512 + w];
  }

  // ---- per-lane persistent state (D layout: vgpr v -> m = v + 8*hi) ----
  const int n = n0 + wv * 16 + l16;       // this lane's global column
  float a = alpha[n];
  a = fminf(fmaxf(a, ALPHA_LO), ALPHA_HI);
  const float one_m_a = 1.f - a;
  float ut[8], st[8];
  for (int v = 0; v < 8; ++v) {
    int b = g * 16 + v + hi * 8;
    ut[v] = ut0[(size_t)b * HDIM + n];
    st[v] = st0[(size_t)b * HDIM + n];
  }
  const int word = slice * 4 + wv;        // this wave's 16-col word in the mask row
  unsigned* myctr = ctr + g;
  const int nl   = wv * 16 + l16;         // local column 0..63
  const int bswz = (nl & 7) << 3;

  auto loadfrag = [&](FragBF& fa, FragBF& fb, int ks) {
    for (int i = 0; i < 8; ++i) fa.u[i] = Asw[ks][lane][i];
    const int bw0 = ((ks * 16 + hi * 4) ^ bswz);
    const int bw1 = ((ks * 16 + 8 + hi * 4) ^ bswz);
    for (int i = 0; i < 4; ++i) { fb.u[i] = Vlds[nl][bw0 + i]; fb.u[4 + i] = Vlds[nl][bw1 + i]; }
  };

  __syncthreads();

  for (int t = 0; t < TSTEPS; ++t) {
    // ---- build WMMA A operand (st tile, bf16) in swizzled LDS layout ----
    if (t == 0) {
      // continuous st0: read fp32 directly, convert
      for (int e = tid; e < 1024; e += 128) {
        int ks = e >> 5, ln2 = e & 31;
        int m2 = ln2 & 15, kh = (ln2 >> 4) * 8;
        const float* src = st0 + (size_t)(g * 16 + m2) * HDIM;
        int kA = ks * 32 + kh, kB = ks * 32 + 16 + kh;
        unsigned* dst = Asw[ks][ln2];
        for (int w2 = 0; w2 < 4; ++w2) {
          dst[w2]     = pack2(src[kA + 2 * w2], src[kA + 2 * w2 + 1]);
          dst[4 + w2] = pack2(src[kB + 2 * w2], src[kB + 2 * w2 + 1]);
        }
      }
    } else {
      const unsigned short* mk = masks + (size_t)(g * 2 + (t & 1)) * 16 * 64;
      for (int e = tid; e < 1024; e += 128) {
        int ks = e >> 5, ln2 = e & 31;
        int m2 = ln2 & 15, sh = (ln2 >> 4) * 8;
        unsigned mw0 = (unsigned)mk[m2 * 64 + 2 * ks]     >> sh;
        unsigned mw1 = (unsigned)mk[m2 * 64 + 2 * ks + 1] >> sh;
        unsigned* dst = Asw[ks][ln2];
        for (int w2 = 0; w2 < 4; ++w2) {
          dst[w2]     = ((mw0 >> (2 * w2)) & 1u) * 0x3F80u
                      | ((mw0 >> (2 * w2 + 1)) & 1u) * 0x3F800000u;
          dst[4 + w2] = ((mw1 >> (2 * w2)) & 1u) * 0x3F80u
                      | ((mw1 >> (2 * w2 + 1)) & 1u) * 0x3F800000u;
        }
      }
    }
    __syncthreads();

    // ---- issue this step's Wx loads early: latency overlaps the WMMAs ----
    float wx[8];
    for (int v = 0; v < 8; ++v) {
      const int b = g * 16 + v + hi * 8;
      const size_t o = ((size_t)b * TSTEPS + t) * HDIM + n;
      wx[v] = Wx[o];
      if (t + 1 < TSTEPS) __builtin_prefetch(&Wx[o + HDIM], 0, 3);
    }

    // ---- r[b,n] = sum_j st[b,j] * Vz[j,n]: 32 WMMAs, 2 accumulators,
    //      fragment loads pipelined one k-step ahead ----
    v8f acc0, acc1;
    for (int i = 0; i < 8; ++i) { acc0[i] = 0.f; acc1[i] = 0.f; }
    FragBF fa0, fb0, fa1, fb1;
    loadfrag(fa0, fb0, 0);
    for (int ks = 0; ks < 32; ks += 2) {
      loadfrag(fa1, fb1, ks + 1);
      acc0 = __builtin_amdgcn_wmma_f32_16x16x32_bf16(
          false, fa0.v, false, fb0.v, (short)0, acc0, false, false);
      if (ks + 2 < 32) loadfrag(fa0, fb0, ks + 2);
      acc1 = __builtin_amdgcn_wmma_f32_16x16x32_bf16(
          false, fa1.v, false, fb1.v, (short)0, acc1, false, false);
    }

    // ---- epilogue: leak/integrate + spike; store output ----
    for (int v = 0; v < 8; ++v) {
      const int b = g * 16 + v + hi * 8;
      const size_t o = ((size_t)b * TSTEPS + t) * HDIM + n;
      float u = a * (ut[v] - st[v]) + one_m_a * (wx[v] + acc0[v] + acc1[v]);
      float s = (u - 1.f > 0.f) ? 1.f : 0.f;
      ut[v] = u; st[v] = s;
      out[o] = s;
    }

    // ---- publish spike bitmask for next step (double-buffered) ----
    {
      unsigned short* mw = masks + (size_t)(g * 2 + ((t + 1) & 1)) * 16 * 64;
      for (int v = 0; v < 8; ++v) {
        unsigned bal = __builtin_amdgcn_ballot_w32(st[v] != 0.f);
        if (lane == 0)  mw[(v)     * 64 + word] = (unsigned short)(bal & 0xFFFFu);
        if (lane == 16) mw[(v + 8) * 64 + word] = (unsigned short)(bal >> 16);
      }
    }

    // ---- group-wide barrier (16 WGs), monotonic counter ----
    __threadfence();                 // make mask stores device-visible
    __syncthreads();
    if (tid == 0) {
      __hip_atomic_fetch_add(myctr, 1u, __ATOMIC_RELEASE, __HIP_MEMORY_SCOPE_AGENT);
      const unsigned target = 16u * (unsigned)(t + 1);
      while (__hip_atomic_load(myctr, __ATOMIC_ACQUIRE, __HIP_MEMORY_SCOPE_AGENT) < target)
        __builtin_amdgcn_s_sleep(2);
    }
    __syncthreads();
    // acquire per thread so every wave's caches see the fresh masks
    (void)__hip_atomic_load(myctr, __ATOMIC_ACQUIRE, __HIP_MEMORY_SCOPE_AGENT);
  }
}

// ---------------------------------------------------------------------------
extern "C" void kernel_launch(void* const* d_in, const int* in_sizes, int n_in,
                              void* d_out, int out_size, void* d_ws, size_t ws_size,
                              hipStream_t stream) {
  const float* x     = (const float*)d_in[0];   // [64,1000,1024]
  const float* W     = (const float*)d_in[1];   // [1024,1024]
  const float* V     = (const float*)d_in[2];   // [1024,1024]
  const float* alpha = (const float*)d_in[3];   // [1024]
  const float* ut0   = (const float*)d_in[4];   // [64,1024]
  const float* st0   = (const float*)d_in[5];   // [64,1024]
  float* out = (float*)d_out;                   // [64,1000,1024]

  char* ws = (char*)d_ws;
  unsigned*       vzt   = (unsigned*)(ws + WS_VZT);
  unsigned short* masks = (unsigned short*)(ws + WS_MASKS);
  unsigned*       ctr   = (unsigned*)(ws + WS_CTR);
  float*          Wx    = (float*)(ws + WS_WX);

  (void)in_sizes; (void)n_in; (void)out_size; (void)ws_size;

  // Phase 0: Vz^T bf16 + barrier counter reset
  rlif_prep<<<(HDIM * 512) / 256, 256, 0, stream>>>(V, vzt, ctr);

  // Phase 1: Wx = x @ W^T   (M = B*T = 64000, N = H = 1024, K = I = 1024)
  dim3 ggrid((BATCH * TSTEPS) / 128, HDIM / 128);
  rlif_gemm<<<ggrid, 256, 0, stream>>>(x, W, Wx);

  // Phase 2: persistent recurrent scan (64 WGs, all resident)
  rlif_scan<<<64, 128, 0, stream>>>(alpha, ut0, st0, Wx, vzt, masks, ctr, out);
}
